// StructuredStateRecurrence_41120016891970
// MI455X (gfx1250) — matrix operand order repaired and checked by reading, not compiled, not run on hardware
//
#include <hip/hip_runtime.h>

// Problem sizes (fixed by the reference)
#define BB 4
#define SS 4096
#define DD 2048
#define MM 512
#define BS (BB * SS)          // 16384 rows
#define CH 16                 // scan chunks
#define CL 256                // chunk length (CH*CL == SS)

typedef __attribute__((ext_vector_type(16))) __bf16 v16bf;
typedef __attribute__((ext_vector_type(8)))  float  v8f;

// 32-byte global->LDS async copy via raw gfx1250 ISA (two b128 ops).
// %0 = VGPR with LDS byte address (low 32 bits of flat shared address),
// %1 = VGPR pair with 64-bit global address. The immediate offset applies to
// both the LDS and global address per the async-load address formula.
#define ASYNC_CP32(gp, lp)                                                     \
  do {                                                                         \
    unsigned _l = (unsigned)(unsigned long long)(lp);                          \
    unsigned long long _g = (unsigned long long)(gp);                          \
    asm volatile("global_load_async_to_lds_b128 %0, %1, off\n\t"               \
                 "global_load_async_to_lds_b128 %0, %1, off offset:16"         \
                 :: "v"(_l), "v"(_g) : "memory");                              \
  } while (0)

#define WAIT_ASYNC() asm volatile("s_wait_asynccnt 0x0" ::: "memory")

__device__ __forceinline__ unsigned short f2bf(float f) {
  unsigned int u = __float_as_uint(f);
  u += 0x7fffu + ((u >> 16) & 1u);   // round-to-nearest-even
  return (unsigned short)(u >> 16);
}

// ---------------------------------------------------------------------------
// Prep: td = 0.9*sigmoid(decay)+0.1, tf = sigmoid(first), tdL = td^256
// ---------------------------------------------------------------------------
__global__ void prep_kernel(const float* __restrict__ tdec,
                            const float* __restrict__ tfir,
                            float* __restrict__ td, float* __restrict__ tf,
                            float* __restrict__ tdL) {
  int m = threadIdx.x;
  float d = 1.f / (1.f + __expf(-tdec[m]));
  float f = 1.f / (1.f + __expf(-tfir[m]));
  float t = d * 0.9f + 0.1f;
  td[m] = t;
  tf[m] = f;
  float p = t;
#pragma unroll
  for (int i = 0; i < 8; ++i) p = p * p;   // td^(2^8) = td^256
  tdL[m] = p;
}

// ---------------------------------------------------------------------------
// f32 -> bf16 (RNE), 4 elements/thread
// ---------------------------------------------------------------------------
__global__ void cvt_bf16_kernel(const float* __restrict__ src,
                                unsigned short* __restrict__ dst, size_t n4) {
  size_t i = (size_t)blockIdx.x * blockDim.x + threadIdx.x;
  if (i >= n4) return;
  float4 f = ((const float4*)src)[i];
  ushort4 o;
  o.x = f2bf(f.x); o.y = f2bf(f.y); o.z = f2bf(f.z); o.w = f2bf(f.w);
  ((ushort4*)dst)[i] = o;
}

// ---------------------------------------------------------------------------
// bf16 WMMA GEMM:  C[row, n] = sum_k A[row, k] * W[n, k]  (+bias, +addSrc)
// A source split at kSplit between A0 (lda0) and A1 (lda1).
// 128x128x32 tiles, 8 waves (2x4), wave tile 64x32 = 4x2 wmma frags.
// Double-buffered LDS fed by async global->LDS copies (ASYNCcnt),
// one barrier per k-step.
// ---------------------------------------------------------------------------
__global__ __launch_bounds__(256) void gemm_bf16_wmma(
    const unsigned short* __restrict__ A0, int lda0,
    const unsigned short* __restrict__ A1, int lda1, int kSplit,
    const unsigned short* __restrict__ W, int ldw,
    const float* __restrict__ bias, const float* __restrict__ addSrc,
    float* __restrict__ C, int N, int K) {
  // 80-byte LDS row pitch (40 ushorts): 16B-aligned, conflict-free b128 reads
  __shared__ unsigned short As[2][128 * 40];
  __shared__ unsigned short Bs[2][128 * 40];

  const int t = threadIdx.x;
  const int lane = t & 31;
  const int wave = t >> 5;
  const int waveM = wave >> 2;   // 0..1 -> 64-row strip
  const int waveN = wave & 3;    // 0..3 -> 32-col strip
  const int m16 = lane & 15;
  const int hf = lane >> 4;

  const int rowBase = blockIdx.x * 128;
  const int colBase = blockIdx.y * 128;

  const int ldr = t >> 1;        // staging: row 0..127
  const int ldh = t & 1;         // staging: 32B half of the 64B row
  const int sOff = ldr * 40 + ldh * 16;   // element offset in LDS tile

  auto aSrc = [&](int kb) -> const unsigned short* {
    return (kb < kSplit)
               ? A0 + (size_t)(rowBase + ldr) * lda0 + kb + ldh * 16
               : A1 + (size_t)(rowBase + ldr) * lda1 + (kb - kSplit) + ldh * 16;
  };
  const unsigned short* bRow = W + (size_t)(colBase + ldr) * ldw + ldh * 16;

  v8f acc[4][2] = {};

  // prologue: tile 0 -> buffer 0
  ASYNC_CP32(aSrc(0), &As[0][sOff]);
  ASYNC_CP32(bRow, &Bs[0][sOff]);

  const int T = K / 32;
  for (int i = 0; i < T; ++i) {
    const int buf = i & 1;
    WAIT_ASYNC();          // tile i resident in LDS (my async ops done)
    __syncthreads();       // everyone's done; prior compute of buf^1 finished
    if (i + 1 < T) {       // prefetch tile i+1 under the WMMAs below
      ASYNC_CP32(aSrc((i + 1) * 32), &As[buf ^ 1][sOff]);
      ASYNC_CP32(bRow + (i + 1) * 32, &Bs[buf ^ 1][sOff]);
    }

    // ---- load fragments per ISA VGPR layouts ----
    union Frag { uint4 q[2]; v16bf v; };
    Frag aF[4], bF[2];
#pragma unroll
    for (int mi = 0; mi < 4; ++mi) {
      int base = (waveM * 64 + mi * 16 + m16) * 40;
      // A row m16: K = {8h..8h+7} then {16+8h..16+8h+7}
      aF[mi].q[0] = *(const uint4*)&As[buf][base + 8 * hf];
      aF[mi].q[1] = *(const uint4*)&As[buf][base + 16 + 8 * hf];
    }
#pragma unroll
    for (int ni = 0; ni < 2; ++ni) {
      // B col m16: K = {8h..8h+7} then {16+8h..16+8h+7}
      int base = (waveN * 32 + ni * 16 + m16) * 40;
      bF[ni].q[0] = *(const uint4*)&Bs[buf][base + 8 * hf];
      bF[ni].q[1] = *(const uint4*)&Bs[buf][base + 16 + 8 * hf];
    }

#pragma unroll
    for (int mi = 0; mi < 4; ++mi)
#pragma unroll
      for (int ni = 0; ni < 2; ++ni)
        acc[mi][ni] = __builtin_amdgcn_wmma_f32_16x16x32_bf16(
            false, aF[mi].v, false, bF[ni].v, (short)0, acc[mi][ni], false,
            false);
  }

  // ---- epilogue: C/D layout -> (M = r + 8*hf, N = m16) per frag ----
#pragma unroll
  for (int mi = 0; mi < 4; ++mi) {
#pragma unroll
    for (int ni = 0; ni < 2; ++ni) {
      int col = colBase + waveN * 32 + ni * 16 + m16;
      float bcol = bias[col];
#pragma unroll
      for (int r = 0; r < 8; ++r) {
        int row = rowBase + waveM * 64 + mi * 16 + hf * 8 + r;
        size_t idx = (size_t)row * N + col;
        float v = acc[mi][ni][r] + bcol;
        if (addSrc) v += addSrc[idx];
        C[idx] = v;
      }
    }
  }
}

// ---------------------------------------------------------------------------
// Scan phase 1: per-chunk local scan, in place over v; emit chunk carries.
// grid = B*CH blocks x M threads
// ---------------------------------------------------------------------------
__global__ void scan_phase1(const float* __restrict__ td, float* __restrict__ v,
                            float* __restrict__ carry) {
  int b = blockIdx.x >> 4;
  int c = blockIdx.x & 15;
  int m = threadIdx.x;
  float tdm = td[m];
  float s = 0.f;
  size_t base = ((size_t)b * SS + (size_t)c * CL) * MM + m;
  for (int i = 0; i < CL; ++i) {
    s = tdm * s + v[base];
    v[base] = s;
    base += MM;
  }
  carry[((size_t)b * CH + c) * MM + m] = s;
}

// ---------------------------------------------------------------------------
// Scan phase 2: serial scan over 16 chunk carries; exclusive carries to E,
// next_memory_state = final state + mem*tf.   grid = B x M threads
// ---------------------------------------------------------------------------
__global__ void scan_phase2(const float* __restrict__ tdL,
                            const float* __restrict__ tf,
                            const float* __restrict__ mem,
                            const float* __restrict__ carry,
                            float* __restrict__ E,
                            float* __restrict__ nextmem) {
  int b = blockIdx.x;
  int m = threadIdx.x;
  float tl = tdL[m];
  float H = 0.f;
  for (int c = 0; c < CH; ++c) {
    size_t idx = ((size_t)b * CH + c) * MM + m;
    E[idx] = H;
    H = tl * H + carry[idx];
  }
  nextmem[b * MM + m] = H + mem[b * MM + m] * tf[m];
}

// ---------------------------------------------------------------------------
// Scan phase 3: wv = local + td^(i+1)*carryIn + mem*tf, emit bf16 for GEMM2.
// grid = B*CH blocks x M threads
// ---------------------------------------------------------------------------
__global__ void scan_phase3(const float* __restrict__ td,
                            const float* __restrict__ tf,
                            const float* __restrict__ mem,
                            const float* __restrict__ E,
                            const float* __restrict__ v,
                            unsigned short* __restrict__ wvbf) {
  int b = blockIdx.x >> 4;
  int c = blockIdx.x & 15;
  int m = threadIdx.x;
  float tdm = td[m];
  float cIn = E[((size_t)b * CH + c) * MM + m];
  float addc = mem[b * MM + m] * tf[m];
  float p = tdm;
  size_t base = ((size_t)b * SS + (size_t)c * CL) * MM + m;
  for (int i = 0; i < CL; ++i) {
    float w = v[base] + p * cIn + addc;
    wvbf[base] = f2bf(w);
    p *= tdm;
    base += MM;
  }
}

// ---------------------------------------------------------------------------
// Launch
// ---------------------------------------------------------------------------
extern "C" void kernel_launch(void* const* d_in, const int* in_sizes, int n_in,
                              void* d_out, int out_size, void* d_ws,
                              size_t ws_size, hipStream_t stream) {
  const float* x    = (const float*)d_in[0];
  const float* mem  = (const float*)d_in[1];
  // d_in[2]=Wk, d_in[3]=bk: dead in the reference (k is never consumed)
  const float* Wv   = (const float*)d_in[4];
  const float* bv   = (const float*)d_in[5];
  const float* Wg   = (const float*)d_in[6];
  const float* bg   = (const float*)d_in[7];
  const float* tdec = (const float*)d_in[8];
  const float* tfir = (const float*)d_in[9];

  float* out = (float*)d_out;                       // (B,S,D)
  float* nextmem = out + (size_t)BS * DD;           // (B,M) appended

  char* ws = (char*)d_ws;
  size_t off = 0;
  auto carve = [&](size_t bytes) -> void* {
    void* p = ws + off;
    off = (off + bytes + 255) & ~(size_t)255;
    return p;
  };
  float* td   = (float*)carve(MM * 4);
  float* tf   = (float*)carve(MM * 4);
  float* tdL  = (float*)carve(MM * 4);
  unsigned short* xbf  = (unsigned short*)carve((size_t)BS * DD * 2);
  unsigned short* wvbf = (unsigned short*)carve((size_t)BS * MM * 2);
  unsigned short* Wvbf = (unsigned short*)carve((size_t)MM * DD * 2);
  unsigned short* Wgbf = (unsigned short*)carve((size_t)DD * (DD + MM) * 2);
  float* vbuf  = (float*)carve((size_t)BS * MM * 4);
  float* carry = (float*)carve((size_t)BB * CH * MM * 4);
  float* Ebuf  = (float*)carve((size_t)BB * CH * MM * 4);

  // 1) scalar prep
  prep_kernel<<<1, MM, 0, stream>>>(tdec, tfir, td, tf, tdL);

  // 2) bf16 conversions
  {
    size_t n4 = (size_t)BS * DD / 4;
    cvt_bf16_kernel<<<(unsigned)(n4 / 256), 256, 0, stream>>>(x, xbf, n4);
    n4 = (size_t)MM * DD / 4;
    cvt_bf16_kernel<<<(unsigned)(n4 / 256), 256, 0, stream>>>(Wv, Wvbf, n4);
    n4 = (size_t)DD * (DD + MM) / 4;
    cvt_bf16_kernel<<<(unsigned)(n4 / 256), 256, 0, stream>>>(Wg, Wgbf, n4);
  }

  // 3) GEMM1: v = x @ Wv^T + bv       (16384 x 512 x 2048)
  {
    dim3 grid(BS / 128, MM / 128);
    gemm_bf16_wmma<<<grid, 256, 0, stream>>>(xbf, DD, xbf, DD, /*kSplit=*/DD,
                                             Wvbf, DD, bv, nullptr, vbuf, MM,
                                             DD);
  }

  // 4) chunked linear scan + memory term, emit wv in bf16
  scan_phase1<<<BB * CH, MM, 0, stream>>>(td, vbuf, carry);
  scan_phase2<<<BB, MM, 0, stream>>>(tdL, tf, mem, carry, Ebuf, nextmem);
  scan_phase3<<<BB * CH, MM, 0, stream>>>(td, tf, mem, Ebuf, vbuf, wvbf);

  // 5) GEMM2: out = x + [x, wv] @ Wg^T + bg   (16384 x 2048 x 2560)
  {
    dim3 grid(BS / 128, DD / 128);
    gemm_bf16_wmma<<<grid, 256, 0, stream>>>(xbf, DD, wvbf, MM,
                                             /*kSplit=*/DD, Wgbf, DD + MM, bg,
                                             x, out, DD, DD + MM);
  }
}
